// AttentionSW_35003983462691
// MI455X (gfx1250) — compile-verified, matrix operand order checked by reading, not run on hardware
//
#include <hip/hip_runtime.h>

// ---------------------------------------------------------------------------
// Sliding-window GQA attention for MI455X (gfx1250, wave32, WMMA).
// v3: GEMM = 128x128 workgroup tiles (4 waves), W k-slice staged in
// double-buffered LDS (20KB of 320KB/WGP), each B fragment feeds 2 WMMAs.
// New in v3: B fragments are software-pipelined out of LDS (fetch chunk c+1
// before issuing chunk c's WMMAs) so the ds_load latency hides under WMMA
// execution instead of an s_wait_dscnt 0 stall per chunk.
// Attention is the flash S^T formulation: lane == query, softmax reductions
// are vgpr-local + one shfl_xor(16), and P re-feeds the PV WMMA with no
// cross-lane transpose.
// ---------------------------------------------------------------------------

#define B_   2
#define L_   2048
#define DIM_ 2048
#define NH_  16
#define NKV_ 8
#define HD_  128
#define WIN_ 128
#define QD_  (NH_ * HD_)   // 2048
#define KD_  (NKV_ * HD_)  // 1024
#define M_   (B_ * L_)     // 4096

typedef __bf16 bf16_t;
typedef __attribute__((ext_vector_type(16))) __bf16 v16bf;
typedef __attribute__((ext_vector_type(8)))  __bf16 v8bf;
typedef __attribute__((ext_vector_type(8)))  float  v8f;

// A-fragment loader for a row-major 16xK32 tile (also used for B via B^T).
// Per CDNA5 ISA 7.12.2 (16-bit A 16x32): lanes 0-15 hold row (lane&15),
// K=0..7 in v0-3 and K=16..23 in v4-7; lanes 16-31 hold K=8..15 / K=24..31.
static __device__ __forceinline__ v16bf load_frag(const bf16_t* p, int stride) {
  const int lane = threadIdx.x & 31;
  const int r    = lane & 15;
  const int cb   = (lane >> 4) << 3;              // +8 cols for upper half-wave
  const bf16_t* q = p + (size_t)r * stride + cb;
  v8bf lo = *reinterpret_cast<const v8bf*>(q);       // K = cb .. cb+7
  v8bf hi = *reinterpret_cast<const v8bf*>(q + 16);  // K = cb+16 .. cb+23
  v16bf out;
#pragma unroll
  for (int i = 0; i < 8; ++i) { out[i] = lo[i]; out[8 + i] = hi[i]; }
  return out;
}

static __device__ __forceinline__ v16bf load_frag_rowclamp(const bf16_t* p, int stride,
                                                           int row0, int maxrow) {
  const int lane = threadIdx.x & 31;
  int r = row0 + (lane & 15);
  r = r < maxrow ? r : maxrow;                    // clamp OOB key rows (masked anyway)
  const int cb = (lane >> 4) << 3;
  const bf16_t* q = p + (size_t)r * stride + cb;
  v8bf lo = *reinterpret_cast<const v8bf*>(q);
  v8bf hi = *reinterpret_cast<const v8bf*>(q + 16);
  v16bf out;
#pragma unroll
  for (int i = 0; i < 8; ++i) { out[i] = lo[i]; out[8 + i] = hi[i]; }
  return out;
}

// ---------------------------------------------------------------------------
// GEMM: C(M,N) = A(M,K) @ W(N,K)^T.
// Workgroup = 4 waves = 128x128 output tile; wave w does rows m0+32w..+31.
// W k-slice (128 rows x 32 k) staged in double-buffered LDS, rows padded to
// 40 bf16 (80B) for conflict-free 16B fragment reads across 64 banks.
// ---------------------------------------------------------------------------
#define LDSS 40  // bf16 elements per LDS row (32 data + 8 pad)

__global__ void __launch_bounds__(128) gemm128x128_bf16(const bf16_t* __restrict__ A,
                                                        const bf16_t* __restrict__ W,
                                                        float* __restrict__ C,
                                                        int K, int N) {
  __shared__ __align__(16) bf16_t bsh[2][128 * LDSS];  // 2 x 10KB
  const int m0   = blockIdx.x * 128;
  const int n0   = blockIdx.y * 128;
  const int tid  = threadIdx.x;
  const int wave = tid >> 5;
  const int lane = tid & 31;

  v8f acc[16] = {};  // [0..7]: rows mbase..+15 ; [8..15]: rows mbase+16..+31

  // Cooperative B stage: thread t copies W row (n0+t), cols k0..k0+31 (64B).
  auto loadB = [&](int k0, int buf) {
    const bf16_t* src = W + (size_t)(n0 + tid) * K + k0;
    v8bf x0 = *reinterpret_cast<const v8bf*>(src);
    v8bf x1 = *reinterpret_cast<const v8bf*>(src + 8);
    v8bf x2 = *reinterpret_cast<const v8bf*>(src + 16);
    v8bf x3 = *reinterpret_cast<const v8bf*>(src + 24);
    bf16_t* dst = &bsh[buf][tid * LDSS];
    *reinterpret_cast<v8bf*>(dst)      = x0;
    *reinterpret_cast<v8bf*>(dst + 8)  = x1;
    *reinterpret_cast<v8bf*>(dst + 16) = x2;
    *reinterpret_cast<v8bf*>(dst + 24) = x3;
  };
  // Fragment read from LDS (same layout rules as load_frag, stride LDSS).
  auto ldsFrag = [&](int buf, int c) -> v16bf {
    const bf16_t* p = &bsh[buf][(16 * c + (lane & 15)) * LDSS + ((lane >> 4) << 3)];
    v8bf lo = *reinterpret_cast<const v8bf*>(p);
    v8bf hi = *reinterpret_cast<const v8bf*>(p + 16);
    v16bf out;
#pragma unroll
    for (int i = 0; i < 8; ++i) { out[i] = lo[i]; out[8 + i] = hi[i]; }
    return out;
  };

  const bf16_t* arow = A + (size_t)(m0 + wave * 32) * K;
  loadB(0, 0);
  int buf = 0;
  for (int k0 = 0; k0 < K; k0 += 32) {
    __syncthreads();                       // bsh[buf] ready; prev reads done
    if (k0 + 32 < K) loadB(k0 + 32, buf ^ 1);
    if (k0 + 64 < K) __builtin_prefetch(arow + k0 + 64, 0, 1);
    v16bf a0 = load_frag(arow + k0, K);
    v16bf a1 = load_frag(arow + (size_t)16 * K + k0, K);
    // Software-pipelined B fragments: issue ds_load for chunk c+1 before
    // the WMMAs of chunk c, hiding LDS latency under the matrix pipe.
    v16bf b = ldsFrag(buf, 0);
#pragma unroll
    for (int c = 0; c < 8; ++c) {
      v16bf bn;
      if (c < 7) bn = ldsFrag(buf, c + 1);
      acc[c]     = __builtin_amdgcn_wmma_f32_16x16x32_bf16(false, a0, false, b,
                                                           (short)0, acc[c], false, false);
      acc[8 + c] = __builtin_amdgcn_wmma_f32_16x16x32_bf16(false, a1, false, b,
                                                           (short)0, acc[8 + c], false, false);
      if (c < 7) b = bn;
    }
    buf ^= 1;
  }
  // C layout: vgpr v -> row m = v + 8*(lane>=16); lane%16 -> col n.
  const int n  = lane & 15;
  const int mh = (lane >> 4) << 3;
  const int mbase = m0 + wave * 32;
#pragma unroll
  for (int g = 0; g < 2; ++g)
#pragma unroll
    for (int c = 0; c < 8; ++c)
#pragma unroll
      for (int v = 0; v < 8; ++v)
        C[(size_t)(mbase + 16 * g + mh + v) * N + n0 + 16 * c + n] = acc[8 * g + c][v];
}

// ---------------------------------------------------------------------------
// Elementwise helpers
// ---------------------------------------------------------------------------
__global__ void f32_to_bf16(const float* __restrict__ in, bf16_t* __restrict__ out, int n) {
  int t = blockIdx.x * blockDim.x + threadIdx.x;
  if (t < n) out[t] = (bf16_t)in[t];
}

// RoPE (HF rotate_half) + cast to bf16. in/out: (B*L, heads*128) row-major.
__global__ void rope_to_bf16(const float* __restrict__ in,
                             const float* __restrict__ cosb,
                             const float* __restrict__ sinb,
                             bf16_t* __restrict__ out, int heads, int total) {
  int t = blockIdx.x * blockDim.x + threadIdx.x;
  if (t >= total) return;
  int d  = t & 63;
  int h  = (t >> 6) % heads;
  int bl = t / (64 * heads);
  int l  = bl & (L_ - 1);
  size_t base = (size_t)bl * heads * HD_ + (size_t)h * HD_;
  float x1 = in[base + d];
  float x2 = in[base + d + 64];
  float c1 = cosb[l * HD_ + d],      s1 = sinb[l * HD_ + d];
  float c2 = cosb[l * HD_ + d + 64], s2 = sinb[l * HD_ + d + 64];
  out[base + d]      = (bf16_t)(x1 * c1 - x2 * s1);
  out[base + d + 64] = (bf16_t)(x2 * c2 + x1 * s2);
}

// V: (B*L, NKV*HD) f32 -> V^T bf16 laid out (B, NKV, HD, L) so the PV WMMA's
// A-fragments (16 dims x 32 keys) load contiguously.
__global__ void v_transpose_bf16(const float* __restrict__ in,
                                 bf16_t* __restrict__ out, int total) {
  int t = blockIdx.x * blockDim.x + threadIdx.x;
  if (t >= total) return;
  int d  = t & (HD_ - 1);
  int hk = (t >> 7) & (NKV_ - 1);
  int bl = t >> 10;                       // / (NKV_*HD_)
  int b  = bl / L_;
  int l  = bl & (L_ - 1);
  out[(((size_t)b * NKV_ + hk) * HD_ + d) * L_ + l] = (bf16_t)in[t];
}

// ---------------------------------------------------------------------------
// Sliding-window flash attention, one wave per (b, h, 16-query tile).
// Computes S^T = K @ Q^T so that lane == query column: softmax reductions are
// 8 vgprs + shfl_xor(16); the S^T C-layout packs directly into the PV WMMA's
// B-fragment (no cross-lane transpose).
// ---------------------------------------------------------------------------
__global__ void attn_sw_kernel(const bf16_t* __restrict__ qr,
                               const bf16_t* __restrict__ kr,
                               const bf16_t* __restrict__ vT,
                               bf16_t* __restrict__ ao) {
  const int r0 = blockIdx.x * 16;
  const int h  = blockIdx.y;
  const int b  = blockIdx.z;
  const int hk = h >> 1;  // N_REP = 2 (jnp.repeat -> head h uses kv head h/2)

  const bf16_t* qbase = qr + ((size_t)b * L_ + r0) * QD_ + (size_t)h * HD_;
  const bf16_t* kbase = kr + (size_t)b * L_ * KD_ + (size_t)hk * HD_;
  const bf16_t* vbase = vT + ((size_t)b * NKV_ + hk) * HD_ * (size_t)L_;

  // Q as the B-matrix (B^T = Q, 16 queries x 128 dims), 4 K-chunks of 32.
  v16bf qf[4];
#pragma unroll
  for (int kc = 0; kc < 4; ++kc) qf[kc] = load_frag(qbase + 32 * kc, QD_);

  v8f o[8] = {};                       // O^T accumulators: 8 dim-chunks x 16 q
  const int lane = threadIdx.x & 31;
  const int hf = lane >> 4;
  const int qn = lane & 15;
  const int i  = r0 + qn;              // this lane's query position
  float mrun = -__builtin_inff();
  float lrun = 0.f;
  const float scale = 0.08838834764831845f;  // 1/sqrt(128)

  int kstart = r0 - WIN_;
  if (kstart < 0) kstart = 0;
  for (int kb = kstart; kb < r0 + 16; kb += 32) {
    // S^T for 32 keys = two 16x16 tiles (keys kb.. and kb+16..)
    v8f sA = {}, sB = {};
#pragma unroll
    for (int kc = 0; kc < 4; ++kc) {
      v16bf ka = load_frag_rowclamp(kbase + 32 * kc, KD_, kb, L_ - 1);
      sA = __builtin_amdgcn_wmma_f32_16x16x32_bf16(false, ka, false, qf[kc],
                                                   (short)0, sA, false, false);
      v16bf kc2 = load_frag_rowclamp(kbase + 32 * kc, KD_, kb + 16, L_ - 1);
      sB = __builtin_amdgcn_wmma_f32_16x16x32_bf16(false, kc2, false, qf[kc],
                                                   (short)0, sB, false, false);
    }
    // mask (causal + window) & scale; S^T row (= key) is v + 8*hf.
    float tmax = -__builtin_inff();
#pragma unroll
    for (int v = 0; v < 8; ++v) {
      int jA = kb + v + 8 * hf;
      int jB = jA + 16;
      bool okA = (jA <= i) && (i - jA <= WIN_);
      bool okB = (jB <= i) && (i - jB <= WIN_);
      sA[v] = okA ? sA[v] * scale : -__builtin_inff();
      sB[v] = okB ? sB[v] * scale : -__builtin_inff();
      tmax = fmaxf(tmax, fmaxf(sA[v], sB[v]));
    }
    tmax = fmaxf(tmax, __shfl_xor(tmax, 16, 32));
    float mnew  = fmaxf(mrun, tmax);     // finite: every block has a live key
    float alpha = __expf(mrun - mnew);
    float rsum  = 0.f;
    v16bf pf;                            // P^T as B-fragment, register-local
#pragma unroll
    for (int v = 0; v < 8; ++v) {
      float pa = __expf(sA[v] - mnew);   // masked -> exactly 0
      float pb = __expf(sB[v] - mnew);
      rsum += pa + pb;
      pf[v]     = (bf16_t)pa;            // keys kb+v(+8*hf)
      pf[8 + v] = (bf16_t)pb;            // keys kb+16+v(+8*hf)
    }
    rsum += __shfl_xor(rsum, 16, 32);
    lrun = alpha * lrun + rsum;
    mrun = mnew;
    // O^T += V^T(chunk) @ P^T ; alpha is per-query == per-lane scalar.
#pragma unroll
    for (int c = 0; c < 8; ++c) {
#pragma unroll
      for (int v = 0; v < 8; ++v) o[c][v] *= alpha;
      v16bf vf = load_frag(vbase + (size_t)(16 * c) * L_ + kb, L_);
      o[c] = __builtin_amdgcn_wmma_f32_16x16x32_bf16(false, vf, false, pf,
                                                     (short)0, o[c], false, false);
    }
  }
  // Normalize + store bf16 to (B*L, NH*HD); per chunk: 8 consecutive dims.
  float inv = 1.f / lrun;
  bf16_t* orow = ao + ((size_t)b * L_ + r0 + qn) * QD_ + (size_t)h * HD_ + 8 * hf;
#pragma unroll
  for (int c = 0; c < 8; ++c) {
    v8bf st;
#pragma unroll
    for (int v = 0; v < 8; ++v) st[v] = (bf16_t)(o[c][v] * inv);
    *reinterpret_cast<v8bf*>(orow + 16 * c) = st;
  }
}

// ---------------------------------------------------------------------------
extern "C" void kernel_launch(void* const* d_in, const int* in_sizes, int n_in,
                              void* d_out, int out_size, void* d_ws, size_t ws_size,
                              hipStream_t stream) {
  (void)in_sizes; (void)n_in; (void)out_size; (void)ws_size;
  const float* x    = (const float*)d_in[0];
  const float* cosb = (const float*)d_in[1];
  const float* sinb = (const float*)d_in[2];
  const float* wq   = (const float*)d_in[3];
  const float* wk   = (const float*)d_in[4];
  const float* wv   = (const float*)d_in[5];
  const float* wo   = (const float*)d_in[6];
  float* out = (float*)d_out;

  char* ws = (char*)d_ws;
  size_t off = 0;
  auto alloc = [&](size_t bytes) -> void* {
    void* p = ws + off;
    off = (off + bytes + 255) & ~(size_t)255;
    return p;
  };
  bf16_t* xb  = (bf16_t*)alloc((size_t)M_ * DIM_ * 2);
  bf16_t* wqb = (bf16_t*)alloc((size_t)QD_ * DIM_ * 2);
  bf16_t* wkb = (bf16_t*)alloc((size_t)KD_ * DIM_ * 2);
  bf16_t* wvb = (bf16_t*)alloc((size_t)KD_ * DIM_ * 2);
  bf16_t* wob = (bf16_t*)alloc((size_t)DIM_ * QD_ * 2);
  float*  qf  = (float*)alloc((size_t)M_ * QD_ * 4);
  float*  kf  = (float*)alloc((size_t)M_ * KD_ * 4);
  float*  vf  = (float*)alloc((size_t)M_ * KD_ * 4);
  bf16_t* qrb = (bf16_t*)alloc((size_t)M_ * QD_ * 2);
  bf16_t* krb = (bf16_t*)alloc((size_t)M_ * KD_ * 2);
  bf16_t* vtb = (bf16_t*)alloc((size_t)M_ * KD_ * 2 + 1024); // +slack for tail reads
  bf16_t* aob = (bf16_t*)alloc((size_t)M_ * QD_ * 2);

  const int EB = 256;
  auto eg = [](int n) { return (n + 255) / 256; };

  // 1) fp32 -> bf16 conversions
  f32_to_bf16<<<eg(M_ * DIM_), EB, 0, stream>>>(x,  xb,  M_ * DIM_);
  f32_to_bf16<<<eg(QD_ * DIM_), EB, 0, stream>>>(wq, wqb, QD_ * DIM_);
  f32_to_bf16<<<eg(KD_ * DIM_), EB, 0, stream>>>(wk, wkb, KD_ * DIM_);
  f32_to_bf16<<<eg(KD_ * DIM_), EB, 0, stream>>>(wv, wvb, KD_ * DIM_);
  f32_to_bf16<<<eg(DIM_ * QD_), EB, 0, stream>>>(wo, wob, DIM_ * QD_);

  // 2) QKV projections (LDS-staged WMMA GEMMs, 128x128 tiles / 4 waves)
  gemm128x128_bf16<<<dim3(M_ / 128, QD_ / 128), 128, 0, stream>>>(xb, wqb, qf, DIM_, QD_);
  gemm128x128_bf16<<<dim3(M_ / 128, KD_ / 128), 128, 0, stream>>>(xb, wkb, kf, DIM_, KD_);
  gemm128x128_bf16<<<dim3(M_ / 128, KD_ / 128), 128, 0, stream>>>(xb, wvb, vf, DIM_, KD_);

  // 3) RoPE (q,k) + V transpose, all cast to bf16
  rope_to_bf16<<<eg(M_ * NH_ * 64),  EB, 0, stream>>>(qf, cosb, sinb, qrb, NH_,  M_ * NH_ * 64);
  rope_to_bf16<<<eg(M_ * NKV_ * 64), EB, 0, stream>>>(kf, cosb, sinb, krb, NKV_, M_ * NKV_ * 64);
  v_transpose_bf16<<<eg(M_ * KD_), EB, 0, stream>>>(vf, vtb, M_ * KD_);

  // 4) sliding-window attention (WMMA flash)
  attn_sw_kernel<<<dim3(L_ / 16, NH_, B_), 32, 0, stream>>>(qrb, krb, vtb, aob);

  // 5) output projection -> fp32 d_out
  gemm128x128_bf16<<<dim3(M_ / 128, DIM_ / 128), 128, 0, stream>>>(aob, wob, out, QD_, DIM_);
}
